// GCN_7043746365666
// MI455X (gfx1250) — matrix operand order repaired and checked by reading, not compiled, not run on hardware
//
#include <hip/hip_runtime.h>
#include <hip/hip_bf16.h>

typedef __attribute__((ext_vector_type(16))) _Float16 v16h;
typedef __attribute__((ext_vector_type(8)))  _Float16 v8h;
typedef __attribute__((ext_vector_type(8)))  float    v8f;

#define H_DIM 64

// ---------------- degree / GCN norm ----------------
__global__ void k_init_deg(float* __restrict__ deg, int n) {
  int i = blockIdx.x * blockDim.x + threadIdx.x;
  if (i < n) deg[i] = 1.0f;               // self-loop contributes 1 to deg[dst]
}

__global__ void k_deg_edges(const int* __restrict__ dst, float* __restrict__ deg, int e) {
  int i = blockIdx.x * blockDim.x + threadIdx.x;
  if (i < e) atomicAdd(&deg[dst[i]], 1.0f);
}

__global__ void k_dinv(float* __restrict__ deg, int n) {
  int i = blockIdx.x * blockDim.x + threadIdx.x;
  if (i < n) deg[i] = rsqrtf(deg[i]);     // deg >= 1 always
}

// ---------------- WMMA GEMM: Y[n x 64] = X[n x K] @ W[K x 64] ----------------
// 128 threads = 4 waves; each wave owns one 16-column tile of the same 16-row tile.
template<int K>
__global__ __launch_bounds__(128) void k_gemm_wmma(const float* __restrict__ X,
                                                   const float* __restrict__ W,
                                                   float* __restrict__ Y,
                                                   int nrows) {
  __shared__ _Float16 ldsA[16 * K];        // row-major [16][K]
  __shared__ _Float16 ldsB[H_DIM * K];     // column-major: ldsB[n*K + k]

  const int tid = threadIdx.x;
  const int m0  = blockIdx.x * 16;
  const bool full = (m0 + 16 <= nrows);    // always true when nrows % 16 == 0

  // Stage A tile (f32 -> f16), coalesced along K
  for (int idx = tid; idx < 16 * K; idx += 128) {
    int row = idx / K, k = idx % K;
    float v = (full || (m0 + row < nrows)) ? X[(size_t)(m0 + row) * K + k] : 0.0f;
    ldsA[idx] = (_Float16)v;
  }
  // Stage W transposed so each lane's B-fragment is contiguous in LDS
  for (int idx = tid; idx < H_DIM * K; idx += 128) {
    int k = idx >> 6, n = idx & 63;        // W is [K][64] row-major
    ldsB[n * K + k] = (_Float16)W[idx];
  }
  __syncthreads();

  const int wv   = tid >> 5;               // n-tile 0..3
  const int lane = tid & 31;
  const int mr   = lane & 15;              // A row within tile
  const int hs   = lane >> 4;              // half-select (K interleave)
  const int ncol = wv * 16 + (lane & 15);  // global output column

  v8f c = {};
  #pragma unroll
  for (int kk = 0; kk < K; kk += 32) {
    // A fragment: lane holds K = kk+hs*8 .. +7  and  kk+16+hs*8 .. +7
    const _Float16* pa = &ldsA[mr * K + kk + hs * 8];
    v8h alo = *(const v8h*)(pa);
    v8h ahi = *(const v8h*)(pa + 16);
    v16h a = __builtin_shufflevector(alo, ahi,
             0,1,2,3,4,5,6,7,8,9,10,11,12,13,14,15);
    // B fragment: column ncol, K = kk+hs*8 .. +7  and  kk+16+hs*8 .. +7
    const _Float16* pb = &ldsB[ncol * K + kk + hs * 8];
    v8h blo = *(const v8h*)(pb);
    v8h bhi = *(const v8h*)(pb + 16);
    v16h b = __builtin_shufflevector(blo, bhi,
             0,1,2,3,4,5,6,7,8,9,10,11,12,13,14,15);
    c = __builtin_amdgcn_wmma_f32_16x16x32_f16(false, a, false, b,
                                               (short)0, c, false, false);
  }

  float* yp = Y + (size_t)(m0 + hs * 8) * H_DIM + ncol;
  if (full) {
    #pragma unroll
    for (int r = 0; r < 8; ++r) yp[(size_t)r * H_DIM] = c[r];   // unconditional
  } else {
    #pragma unroll
    for (int r = 0; r < 8; ++r) {
      int row = m0 + r + hs * 8;
      if (row < nrows) Y[(size_t)row * H_DIM + ncol] = c[r];
    }
  }
}

// ---------------- aggregation ----------------
// h[i,f] = xw[i,f] * dinv[i]^2   (self-loop term; also zero-initializes h)
// float4 vectorized: 16 threads per node
__global__ void k_selfloop(const float4* __restrict__ xw, const float* __restrict__ dinv,
                           float4* __restrict__ h, int n) {
  int idx = blockIdx.x * blockDim.x + threadIdx.x;
  if (idx < n * (H_DIM / 4)) {
    int i = idx >> 4;
    float w = dinv[i]; w *= w;
    float4 v = xw[idx];
    v.x *= w; v.y *= w; v.z *= w; v.w *= w;
    h[idx] = v;
  }
}

// h[dst,f] += xw[src,f] * dinv[src]*dinv[dst]   (one wave per edge, float2/lane)
__global__ void k_scatter(const int* __restrict__ src, const int* __restrict__ dst,
                          const float* __restrict__ dinv, const float2* __restrict__ xw,
                          float* __restrict__ h, int e) {
  long long idx = (long long)blockIdx.x * blockDim.x + threadIdx.x;
  if (idx >= (long long)e * (H_DIM / 2)) return;
  int ei = (int)(idx >> 5);        // edge (uniform per wave)
  int f2 = (int)(idx & 31);        // float2 slot
  int s = src[ei], d = dst[ei];
  float w = dinv[s] * dinv[d];
  float2 v = xw[(size_t)s * (H_DIM / 2) + f2];
  float* hp = &h[(size_t)d * H_DIM + 2 * f2];
  atomicAdd(hp,     v.x * w);
  atomicAdd(hp + 1, v.y * w);
}

__global__ void k_bias_relu(float4* __restrict__ h, const float4* __restrict__ b, int n) {
  int idx = blockIdx.x * blockDim.x + threadIdx.x;
  if (idx < n * (H_DIM / 4)) {
    float4 bb = b[idx & 15];
    float4 v = h[idx];
    v.x = fmaxf(v.x + bb.x, 0.0f);
    v.y = fmaxf(v.y + bb.y, 0.0f);
    v.z = fmaxf(v.z + bb.z, 0.0f);
    v.w = fmaxf(v.w + bb.w, 0.0f);
    h[idx] = v;
  }
}

// ---------------- pooling + head ----------------
__global__ void k_pool_zero(float* __restrict__ sums, float* __restrict__ cnt, int g) {
  int idx = blockIdx.x * blockDim.x + threadIdx.x;
  if (idx < g * H_DIM) sums[idx] = 0.0f;
  if (idx < g) cnt[idx] = 0.0f;
}

__global__ void k_pool(const float4* __restrict__ h, const int* __restrict__ batch,
                       float* __restrict__ sums, float* __restrict__ cnt, int n) {
  int idx = blockIdx.x * blockDim.x + threadIdx.x;
  if (idx < n * (H_DIM / 4)) {
    int i = idx >> 4, fq = idx & 15;
    int g = batch[i];
    float4 v = h[idx];
    float* sp = &sums[(size_t)g * H_DIM + 4 * fq];
    atomicAdd(sp,     v.x);
    atomicAdd(sp + 1, v.y);
    atomicAdd(sp + 2, v.z);
    atomicAdd(sp + 3, v.w);
    if (fq == 0) atomicAdd(&cnt[g], 1.0f);
  }
}

__global__ void k_head(const float* __restrict__ sums, const float* __restrict__ cnt,
                       const float* __restrict__ Wl, const float* __restrict__ bl,
                       float* __restrict__ out, int g) {
  int i = blockIdx.x * blockDim.x + threadIdx.x;
  if (i < g) {
    float inv = 1.0f / fmaxf(cnt[i], 1.0f);
    float acc = bl[0];
    #pragma unroll
    for (int f = 0; f < H_DIM; ++f)
      acc += sums[(size_t)i * H_DIM + f] * inv * Wl[f];
    out[i] = 1.0f / (1.0f + __expf(-acc));
  }
}

// ---------------- launcher ----------------
extern "C" void kernel_launch(void* const* d_in, const int* in_sizes, int n_in,
                              void* d_out, int out_size, void* d_ws, size_t ws_size,
                              hipStream_t stream) {
  const float* x     = (const float*)d_in[0];
  const int*   eidx  = (const int*)d_in[1];
  const int*   batch = (const int*)d_in[2];
  const float* W1    = (const float*)d_in[3];
  const float* b1    = (const float*)d_in[4];
  const float* W2    = (const float*)d_in[5];
  const float* b2    = (const float*)d_in[6];
  const float* Wl    = (const float*)d_in[7];
  const float* bl    = (const float*)d_in[8];
  float* out = (float*)d_out;

  const int n = in_sizes[0] / 128;   // nodes
  const int e = in_sizes[1] / 2;     // edges (no self-loops)
  const int g = out_size;            // graphs

  const int* srcp = eidx;
  const int* dstp = eidx + e;

  // workspace layout (floats): dinv | bufA (xw) | bufB (h) | sums | cnt
  float* ws   = (float*)d_ws;
  float* dinv = ws;
  float* bufA = dinv + n;
  float* bufB = bufA + (size_t)n * H_DIM;
  float* sums = bufB + (size_t)n * H_DIM;
  float* cnt  = sums + (size_t)g * H_DIM;

  const int B = 256;
  const int nb_n  = (n + B - 1) / B;
  const int nb_e  = (e + B - 1) / B;
  const int nb_q  = (n * (H_DIM / 4) + B - 1) / B;                 // float4 grids
  const long long e2 = (long long)e * (H_DIM / 2);
  const int nb_e2 = (int)((e2 + B - 1) / B);                       // scatter grid
  const int mtiles = (n + 15) / 16;

  // GCN normalization
  k_init_deg <<<nb_n, B, 0, stream>>>(dinv, n);
  k_deg_edges<<<nb_e, B, 0, stream>>>(dstp, dinv, e);
  k_dinv     <<<nb_n, B, 0, stream>>>(dinv, n);

  // ----- layer 1: xw1 = x @ W1 (WMMA), aggregate, bias+relu -----
  k_gemm_wmma<128><<<mtiles, 128, 0, stream>>>(x, W1, bufA, n);
  k_selfloop <<<nb_q,  B, 0, stream>>>((const float4*)bufA, dinv, (float4*)bufB, n);
  k_scatter  <<<nb_e2, B, 0, stream>>>(srcp, dstp, dinv, (const float2*)bufA, bufB, e);
  k_bias_relu<<<nb_q,  B, 0, stream>>>((float4*)bufB, (const float4*)b1, n);

  // ----- layer 2: xw2 = h1 @ W2 (WMMA), aggregate, bias+relu -----
  k_gemm_wmma<64><<<mtiles, 128, 0, stream>>>(bufB, W2, bufA, n);
  k_selfloop <<<nb_q,  B, 0, stream>>>((const float4*)bufA, dinv, (float4*)bufB, n);
  k_scatter  <<<nb_e2, B, 0, stream>>>(srcp, dstp, dinv, (const float2*)bufA, bufB, e);
  k_bias_relu<<<nb_q,  B, 0, stream>>>((float4*)bufB, (const float4*)b2, n);

  // ----- global mean pool + linear + sigmoid -----
  k_pool_zero<<<(g * H_DIM + B - 1) / B, B, 0, stream>>>(sums, cnt, g);
  k_pool     <<<nb_q, B, 0, stream>>>((const float4*)bufB, batch, sums, cnt, n);
  k_head     <<<(g + B - 1) / B, B, 0, stream>>>(sums, cnt, Wl, bl, out, g);
}
